// DiagnosticRIN_75892072120992
// MI455X (gfx1250) — matrix operand order repaired and checked by reading, not compiled, not run on hardware
//
#include <hip/hip_runtime.h>
#include <stdint.h>

typedef __attribute__((ext_vector_type(16))) _Float16 v16h;
typedef __attribute__((ext_vector_type(8)))  _Float16 v8h;
typedef __attribute__((ext_vector_type(4)))  _Float16 v4h;
typedef __attribute__((ext_vector_type(8)))  float    v8f;

#define LUTN     4096
#define TWO_PI_F 6.28318530717958647692f
#define PHI_F    1.61803398874989484820f
#define D_HALF   256
#define KDIM     512      // 2*D
#define NDIM     32000    // V
#define SLEN     2048
#define BNUM     4

// ---------------- Kernel 1: proj_weight f32 -> f16 ----------------
__global__ void k_cvt_f16(const float* __restrict__ src, _Float16* __restrict__ dst, int n4) {
  int i = blockIdx.x * blockDim.x + threadIdx.x;
  if (i >= n4) return;
  float4 v = ((const float4*)src)[i];
  v4h h;
  h[0] = (_Float16)v.x; h[1] = (_Float16)v.y;
  h[2] = (_Float16)v.z; h[3] = (_Float16)v.w;
  ((v4h*)dst)[i] = h;
}

// ---------------- Kernel 2: gather + sequential RIN scan ----------------
__global__ void k_scan(const int* __restrict__ ids,
                       const float* __restrict__ embw,
                       _Float16* __restrict__ comb) {
  __shared__ float s_sin[LUTN];
  __shared__ float s_cos[LUTN];
  const int d = threadIdx.x;   // 0..255
  const int b = blockIdx.x;    // 0..3
  for (int i = d; i < LUTN; i += blockDim.x) {
    float ang = (float)i * (TWO_PI_F / (float)LUTN);
    s_sin[i] = sinf(ang);
    s_cos[i] = cosf(ang);
  }
  __syncthreads();

  const float sc = (float)LUTN / TWO_PI_F;
  float hr = 0.0f, hi = 0.0f;
  const int* idrow = ids + (size_t)b * SLEN;
  for (int t = 0; t < SLEN; ++t) {
    int id = idrow[t];
    const float* e = embw + (size_t)id * KDIM;
    float w  = e[d];
    float bb = e[D_HALF + d];
    float inv = 1.0f / (1.0f + fabsf(w));
    float tp  = (float)t * PHI_F;
    float thr = hr * inv + bb + tp;
    float thi = hi * inv + bb + tp;
    int ir = ((int)floorf(thr * sc)) & (LUTN - 1);
    int ii = ((int)floorf(thi * sc)) & (LUTN - 1);
    float sr = s_sin[ir], cr = s_cos[ir];
    float si = s_sin[ii], ci = s_cos[ii];
    float nr = cr * ci - sr * si;
    float ni = cr * si + sr * ci;
    hr = nr; hi = ni;
    size_t o = ((size_t)b * SLEN + t) * KDIM;
    comb[o + d]          = (_Float16)nr;
    comb[o + D_HALF + d] = (_Float16)ni;
  }
}

// -------- async 16B global -> LDS copy (gfx1250, tracked by ASYNCcnt) --------
__device__ __forceinline__ void async_cp16(uint32_t lds_addr, uint64_t gaddr) {
  asm volatile("global_load_async_to_lds_b128 %0, %1, off"
               :: "v"(lds_addr), "v"(gaddr) : "memory");
}
__device__ __forceinline__ void wait_async0() {
  asm volatile("s_wait_asynccnt 0x0" ::: "memory");
}
__device__ __forceinline__ uint32_t lds_addr_of(const void* p) {
  // generic shared-aperture pointer: LDS byte address = addr[31:0]
  return (uint32_t)(uintptr_t)p;
}

// ---------------- Kernel 3: WMMA GEMM + bias, async-LDS double buffered ------
// C[M,N] = A[M,K] * Bw[N,K]^T + bias, M=8192, K=512, N=32000.
// Block: 256 thr = 8 waves (4 M-rows x 2 N-cols), block tile 64x128,
// wave tile 16x64. 16 K-stages of 32; per stage the block stages A(4KB)+B(8KB)
// into LDS via 3 async b128 copies per thread, double buffered. Output stores
// are non-temporal so the 1.05 GB logit stream doesn't evict the L2-resident
// A/B operands (A 8.4MB + B 32.8MB << 192MB L2).
__global__ void __launch_bounds__(256, 2)
k_gemm(const _Float16* __restrict__ A,   // [M, K] f16 row-major
       const _Float16* __restrict__ Bw,  // [N, K] f16 row-major
       const float*    __restrict__ bias,// [N]
       float*          __restrict__ C) { // [M, N] f32
  __shared__ _Float16 sA[2][64 * 32];    //  8 KB
  __shared__ _Float16 sB[2][128 * 32];   // 16 KB

  const int tid  = threadIdx.x;
  const int lane = tid & 31;
  const int wave = tid >> 5;           // 0..7
  const int wrow = wave >> 1;          // 0..3 (M)
  const int wcol = wave & 1;           // 0..1 (N)
  const int l16  = lane & 15;
  const int lhi  = lane >> 4;          // 0/1

  const int mBlock = blockIdx.y * 64;
  const int nBlock = blockIdx.x * 128;

  // async-copy work assignment: each thread moves 1 A chunk + 2 B chunks (16B each)
  const int row  = tid >> 2;           // 0..63
  const int chk  = tid & 3;            // 0..3 (which 8-half chunk of the 32-K slab)
  const uint64_t gA  = (uint64_t)(A  + (size_t)(mBlock + row)      * KDIM + chk * 8);
  const uint64_t gB0 = (uint64_t)(Bw + (size_t)(nBlock + row)      * KDIM + chk * 8);
  const uint64_t gB1 = (uint64_t)(Bw + (size_t)(nBlock + 64 + row) * KDIM + chk * 8);
  uint32_t lA[2], lB0[2], lB1[2];
#pragma unroll
  for (int p = 0; p < 2; ++p) {
    lA[p]  = lds_addr_of(&sA[p][row * 32 + chk * 8]);
    lB0[p] = lds_addr_of(&sB[p][row * 32 + chk * 8]);
    lB1[p] = lds_addr_of(&sB[p][(64 + row) * 32 + chk * 8]);
  }

  v8f acc0 = {}, acc1 = {}, acc2 = {}, acc3 = {};

  // prologue: stage 0 into buffer 0
  async_cp16(lA[0],  gA);
  async_cp16(lB0[0], gB0);
  async_cp16(lB1[0], gB1);

  const int NSTAGE = KDIM / 32; // 16
#pragma unroll 2
  for (int s = 0; s < NSTAGE; ++s) {
    const int p = s & 1;
    wait_async0();
    __syncthreads();
    if (s + 1 < NSTAGE) {
      const size_t kb = (size_t)(s + 1) * 32 * sizeof(_Float16); // byte offset
      async_cp16(lA[p ^ 1],  gA  + kb);
      async_cp16(lB0[p ^ 1], gB0 + kb);
      async_cp16(lB1[p ^ 1], gB1 + kb);
    }
    // A fragment: row m = l16, 16 contiguous K at lhi*16
    const _Float16* pa = &sA[p][(wrow * 16 + l16) * 32];
    v8h alo = *(const v8h*)(pa + lhi * 16);
    v8h ahi = *(const v8h*)(pa + lhi * 16 + 8);
    v16h a  = __builtin_shufflevector(alo, ahi,
                                      0,1,2,3,4,5,6,7,8,9,10,11,12,13,14,15);
    // B fragments: col n = wcol*64 + j*16 + l16, 16 contiguous K at lhi*16
    v16h f[4];
#pragma unroll
    for (int j = 0; j < 4; ++j) {
      f[j] = *(const v16h*)(&sB[p][(wcol * 64 + j * 16 + l16) * 32 + lhi * 16]);
    }
    acc0 = __builtin_amdgcn_wmma_f32_16x16x32_f16(false, a, false, f[0], (short)0, acc0, false, false);
    acc1 = __builtin_amdgcn_wmma_f32_16x16x32_f16(false, a, false, f[1], (short)0, acc1, false, false);
    acc2 = __builtin_amdgcn_wmma_f32_16x16x32_f16(false, a, false, f[2], (short)0, acc2, false, false);
    acc3 = __builtin_amdgcn_wmma_f32_16x16x32_f16(false, a, false, f[3], (short)0, acc3, false, false);
  }

  // Epilogue: C/D layout => VGPR r holds row (r + 8*lhi), column l16; add bias.
  // Non-temporal stores: logits are write-once, keep L2 for A/B tiles.
  const int rbase = mBlock + wrow * 16 + 8 * lhi;
  const int nWave = nBlock + wcol * 64;
#pragma unroll
  for (int j = 0; j < 4; ++j) {
    v8f acc = (j == 0) ? acc0 : (j == 1) ? acc1 : (j == 2) ? acc2 : acc3;
    int n = nWave + j * 16 + l16;
    float bv = bias[n];
    float* cp = C + (size_t)rbase * NDIM + n;
#pragma unroll
    for (int r = 0; r < 8; ++r) {
      __builtin_nontemporal_store(acc[r] + bv, cp + (size_t)r * NDIM);
    }
  }
}

extern "C" void kernel_launch(void* const* d_in, const int* in_sizes, int n_in,
                              void* d_out, int out_size, void* d_ws, size_t ws_size,
                              hipStream_t stream) {
  const int*   ids   = (const int*)d_in[0];    // (B, S) int32
  const float* embw  = (const float*)d_in[1];  // (V, 2D) f32
  const float* projw = (const float*)d_in[2];  // (V, 2D) f32
  const float* pbias = (const float*)d_in[3];  // (V,)    f32
  float* out = (float*)d_out;                  // (B, S, V) f32

  _Float16* comb = (_Float16*)d_ws;
  size_t combBytes = (size_t)BNUM * SLEN * KDIM * sizeof(_Float16); // 8.4 MB
  _Float16* pw16 = (_Float16*)((char*)d_ws + combBytes);            // 32.8 MB

  int n4 = (NDIM * KDIM) / 4;
  k_cvt_f16<<<(n4 + 255) / 256, 256, 0, stream>>>(projw, pw16, n4);
  k_scan<<<BNUM, D_HALF, 0, stream>>>(ids, embw, comb);

  dim3 grid(NDIM / 128, (BNUM * SLEN) / 64); // (250, 128)
  k_gemm<<<grid, 256, 0, stream>>>(comb, pw16, pbias, out);

  (void)in_sizes; (void)n_in; (void)out_size; (void)ws_size;
}
